// QKVAttention_9079560864530
// MI455X (gfx1250) — matrix operand order repaired and checked by reading, not compiled, not run on hardware
//
#include <hip/hip_runtime.h>
#include <hip/hip_bf16.h>

// ---------------------------------------------------------------------------
// Fused flash-attention for the QKVAttention reference, targeting gfx1250.
// qkv: [4,3072,2048] f32, encoder_kv: [4,2048,256] f32 -> out [4,1024,2048] f32
// 64 batch-heads, ch=64, T=2048, S=2304. bf16 WMMA (16x16x32) + fp32 accum.
// 256 threads = 8 wave32s per workgroup; each workgroup owns a 128-query tile.
// ---------------------------------------------------------------------------

typedef __attribute__((ext_vector_type(16))) __bf16 v16bf;
typedef __attribute__((ext_vector_type(8)))  float  v8f;

#define CH     64
#define STILE  64
#define TTILE  128
#define NWAVE  8
#define T_LEN  2048
#define SENC   256
#define STOT   2304
#define LSTR   72     // padded LDS row stride (bf16 elements)

// Native bf16 converts (v_cvt_pk_bf16_f32 on gfx1250) instead of manual RNE.
__device__ __forceinline__ unsigned short bf1(float a) {
    union { __bf16 h; unsigned short u; } p; p.h = (__bf16)a; return p.u;
}
__device__ __forceinline__ unsigned bf2(float a, float b) {
    union { __bf16 h[2]; unsigned u; } p;
    p.h[0] = (__bf16)a; p.h[1] = (__bf16)b; return p.u;
}

union AB { v16bf v; unsigned u[8]; };

__global__ __launch_bounds__(256)
void qkv_attn_kernel(const float* __restrict__ qkv,
                     const float* __restrict__ ekv,
                     float* __restrict__ out) {
    __shared__ unsigned short Qt[TTILE * LSTR];  // Q transposed: [t_local][c]
    __shared__ unsigned short Kt[STILE * LSTR];  // K transposed: [s_local][c]
    __shared__ unsigned short Vn[CH    * LSTR];  // V natural:    [c][s_local]
    __shared__ unsigned short Pl[TTILE * LSTR];  // P (bf16):     [t_local][s_local]

    const int tid  = threadIdx.x;
    const int lane = tid & 31;          // wave32 (gfx1250)
    const int wave = tid >> 5;          // 8 waves
    const int l16  = lane & 15;
    const int ah   = (lane >> 4) & 1;   // lane half-group

    const int ttile = blockIdx.x;       // 0..15
    const int bh    = blockIdx.y;       // 0..63
    const int b = bh >> 4, h = bh & 15;
    const int t0 = ttile * TTILE;

    const float* qbase  = qkv + (size_t)(b * 3072 + h * 192      ) * T_LEN;
    const float* kqbase = qkv + (size_t)(b * 3072 + h * 192 +  64) * T_LEN;
    const float* vqbase = qkv + (size_t)(b * 3072 + h * 192 + 128) * T_LEN;
    const float* ekbase = ekv + (size_t)(b * 2048 + h * 128      ) * SENC;
    const float* evbase = ekv + (size_t)(b * 2048 + h * 128 +  64) * SENC;
    float*       obase  = out + (size_t)(b * 1024 + h *  64      ) * T_LEN;

    const float qscale = 0.125f;        // ch^-0.5 (both ch^-0.25 factors) folded into Q

    // ---- stage Q tile transposed (coalesced f32x4 reads along t) ----
    // 128 t x 64 c = 2048 float4-reads = 8 per thread
#pragma unroll
    for (int r = 0; r < 8; ++r) {
        int idx = tid + 256 * r;        // 0..2047
        int c   = idx >> 5;             // 0..63
        int q4  = idx & 31;             // 0..31 (t_local = q4*4)
        const float4 f = *(const float4*)(qbase + (size_t)c * T_LEN + t0 + q4 * 4);
        Qt[(q4 * 4 + 0) * LSTR + c] = bf1(f.x * qscale);
        Qt[(q4 * 4 + 1) * LSTR + c] = bf1(f.y * qscale);
        Qt[(q4 * 4 + 2) * LSTR + c] = bf1(f.z * qscale);
        Qt[(q4 * 4 + 3) * LSTR + c] = bf1(f.w * qscale);
    }
    __syncthreads();

    // ---- Q A-fragments (ISA 16-bit A 16x32 layout): row m = lane%16 ----
    const int am = wave * 16 + l16;     // t row owned by this lane as A-row
    AB aq[2];
#pragma unroll
    for (int kb = 0; kb < 2; ++kb)
#pragma unroll
        for (int j = 0; j < 8; ++j) {
            int k = kb * 32 + ((j >> 2) << 4) + (ah << 3) + ((j & 3) << 1);
            aq[kb].u[j] = *(const unsigned*)(&Qt[am * LSTR + k]);
        }

    // ---- accumulators + online-softmax state (rows m = i + 8*ah) ----
    v8f acc[4];
#pragma unroll
    for (int nt = 0; nt < 4; ++nt)
#pragma unroll
        for (int e = 0; e < 8; ++e) acc[nt][e] = 0.0f;
    float rm[8], rl[8];
#pragma unroll
    for (int i = 0; i < 8; ++i) { rm[i] = -INFINITY; rl[i] = 0.0f; }

    for (int st = 0; st < STOT / STILE; ++st) {
        __syncthreads();                           // prev iter done reading Kt/Vn
        const bool   enc  = (st < SENC / STILE);   // tiles 0..3 from encoder_kv
        const int    scol = enc ? st * STILE : st * STILE - SENC;
        const float* ks   = enc ? ekbase : kqbase;
        const float* vs   = enc ? evbase : vqbase;
        const size_t kstr = enc ? (size_t)SENC : (size_t)T_LEN;
        const bool   pf   = (enc && st < 3) || (!enc && st < STOT / STILE - 1);

        // 64 c x 64 s tile = 1024 float4-reads each for K and V = 4 per thread
#pragma unroll
        for (int r = 0; r < 4; ++r) {
            int idx = tid + 256 * r;
            int c   = idx >> 4;
            int q4  = idx & 15;
            const float4 fk = *(const float4*)(ks + (size_t)c * kstr + scol + q4 * 4);
            Kt[(q4 * 4 + 0) * LSTR + c] = bf1(fk.x);   // transpose into LDS
            Kt[(q4 * 4 + 1) * LSTR + c] = bf1(fk.y);
            Kt[(q4 * 4 + 2) * LSTR + c] = bf1(fk.z);
            Kt[(q4 * 4 + 3) * LSTR + c] = bf1(fk.w);
            const float4 fv = *(const float4*)(vs + (size_t)c * kstr + scol + q4 * 4);
            *(unsigned*)(&Vn[c * LSTR + q4 * 4    ]) = bf2(fv.x, fv.y);
            *(unsigned*)(&Vn[c * LSTR + q4 * 4 + 2]) = bf2(fv.z, fv.w);
            if (pf) {                                  // gfx1250 global_prefetch_b8
                __builtin_prefetch(ks + (size_t)c * kstr + scol + STILE + q4 * 4, 0, 1);
                __builtin_prefetch(vs + (size_t)c * kstr + scol + STILE + q4 * 4, 0, 1);
            }
        }
        __syncthreads();

        // ---- scores S = Q^T K  (16x64 per wave, fp32 accum) ----
        v8f sc[4];
#pragma unroll
        for (int nt = 0; nt < 4; ++nt) {
            v8f s;
#pragma unroll
            for (int e = 0; e < 8; ++e) s[e] = 0.0f;
#pragma unroll
            for (int kb = 0; kb < 2; ++kb) {
                AB bk;  // B 32x16: VGPR j <- rows k=16*(j>>2)+8*ah+2*(j&3),+1 ; col n=lane%16
#pragma unroll
                for (int j = 0; j < 8; ++j) {
                    int k = kb * 32 + ((j >> 2) << 4) + (ah << 3) + ((j & 3) << 1);
                    bk.u[j] = *(const unsigned*)(&Kt[(nt * 16 + l16) * LSTR + k]);
                }
                s = __builtin_amdgcn_wmma_f32_16x16x32_bf16(
                        false, aq[kb].v, false, bk.v, (short)0, s, false, false);
            }
            sc[nt] = s;
        }

        // ---- online softmax: D-layout row m = i + 8*ah, cols across 16 lanes ----
#pragma unroll
        for (int i = 0; i < 8; ++i) {
            float mx = sc[0][i];
#pragma unroll
            for (int nt = 1; nt < 4; ++nt) mx = fmaxf(mx, sc[nt][i]);
#pragma unroll
            for (int msk = 1; msk < 16; msk <<= 1)
                mx = fmaxf(mx, __shfl_xor(mx, msk, 32));
            float mnew = fmaxf(rm[i], mx);
            float corr = __expf(rm[i] - mnew);
            rm[i] = mnew;
            float ps = 0.0f;
#pragma unroll
            for (int nt = 0; nt < 4; ++nt) {
                float p = __expf(sc[nt][i] - mnew);
                ps += p;
                sc[nt][i] = p;
                acc[nt][i] *= corr;
            }
#pragma unroll
            for (int msk = 1; msk < 16; msk <<= 1)
                ps += __shfl_xor(ps, msk, 32);
            rl[i] = rl[i] * corr + ps;
            int trow = wave * 16 + i + 8 * ah;        // wave-private Pl strip
#pragma unroll
            for (int nt = 0; nt < 4; ++nt)
                Pl[trow * LSTR + nt * 16 + l16] = bf1(sc[nt][i]);
        }
        // Pl strip is written & read only by this wave: DS ops are in-order,
        // backend inserts s_wait_dscnt — no barrier needed here.

        // ---- O += P x V^T ----
#pragma unroll
        for (int sb = 0; sb < 2; ++sb) {
            AB ap;
#pragma unroll
            for (int j = 0; j < 8; ++j) {
                int k = sb * 32 + ((j >> 2) << 4) + (ah << 3) + ((j & 3) << 1);
                ap.u[j] = *(const unsigned*)(&Pl[am * LSTR + k]);
            }
#pragma unroll
            for (int nt = 0; nt < 4; ++nt) {
                AB bv;  // B[k=s][n=c] = Vn[c][s]: pair along s contiguous
#pragma unroll
                for (int j = 0; j < 8; ++j) {
                    int k = sb * 32 + ((j >> 2) << 4) + (ah << 3) + ((j & 3) << 1);
                    bv.u[j] = *(const unsigned*)(&Vn[(nt * 16 + l16) * LSTR + k]);
                }
                acc[nt] = __builtin_amdgcn_wmma_f32_16x16x32_bf16(
                              false, ap.v, false, bv.v, (short)0, acc[nt], false, false);
            }
        }
    }

    // ---- epilogue: normalize by row sum, store [c][t] ----
#pragma unroll
    for (int i = 0; i < 8; ++i) {
        float inv = 1.0f / rl[i];
        int t = t0 + wave * 16 + i + 8 * ah;
#pragma unroll
        for (int nt = 0; nt < 4; ++nt) {
            int c = nt * 16 + l16;
            obase[(size_t)c * T_LEN + t] = acc[nt][i] * inv;
        }
    }
}

extern "C" void kernel_launch(void* const* d_in, const int* in_sizes, int n_in,
                              void* d_out, int out_size, void* d_ws, size_t ws_size,
                              hipStream_t stream) {
    (void)in_sizes; (void)n_in; (void)d_ws; (void)ws_size; (void)out_size;
    const float* qkv = (const float*)d_in[0];
    const float* ekv = (const float*)d_in[1];
    float* out = (float*)d_out;
    dim3 grid(T_LEN / TTILE, 64);   // 16 query tiles x 64 batch-heads
    dim3 block(256);                // 8 wave32s
    qkv_attn_kernel<<<grid, block, 0, stream>>>(qkv, ekv, out);
}
